// MCBD_2628519985587
// MI455X (gfx1250) — matrix operand order repaired and checked by reading, not compiled, not run on hardware
//
#include <hip/hip_runtime.h>
#include <hip/hip_bf16.h>

// ---------- types ----------
typedef __attribute__((ext_vector_type(16))) __bf16 v16bf;
typedef __attribute__((ext_vector_type(8)))  __bf16 v8bf;
typedef __attribute__((ext_vector_type(8)))  float  v8f;
typedef __attribute__((ext_vector_type(4)))  unsigned int u32x4;
typedef __attribute__((ext_vector_type(8)))  int i32x8;
typedef __attribute__((ext_vector_type(4)))  int i32x4;

__device__ __forceinline__ __bf16 f32_to_bf16(float f) {
  union { float f; unsigned u; } v; v.f = f;
  unsigned r = v.u + 0x7FFFu + ((v.u >> 16) & 1u);   // round-to-nearest-even
  unsigned short h = (unsigned short)(r >> 16);
  __bf16 out; __builtin_memcpy(&out, &h, 2); return out;
}
__device__ __forceinline__ float bf16_to_f32(__bf16 b) {
  unsigned short h; __builtin_memcpy(&h, &b, 2);
  union { unsigned u; float f; } v; v.u = ((unsigned)h) << 16; return v.f;
}
__device__ __forceinline__ float sigmoidf(float z) {
  return 1.0f / (1.0f + __expf(-z));
}

// ---------- elementwise f32 -> bf16 cast ----------
__global__ void k_cast_bf16(const float* __restrict__ X, __bf16* __restrict__ Y, int n) {
  int idx = blockIdx.x * 256 + threadIdx.x;
  if (idx < n) Y[idx] = f32_to_bf16(X[idx]);
}

// ---------- direct conv1d (k=3, pad=1) + ReLU ----------
__global__ void k_conv1d_relu(const float* __restrict__ X, int cinTotal, int cin,
                              const float* __restrict__ W, const float* __restrict__ bias,
                              float* __restrict__ Y, int cout) {
  const int T = 128, B = 4;
  int idx = blockIdx.x * 256 + threadIdx.x;          // over B*cout*T
  if (idx >= B * cout * T) return;
  int t  = idx % T;
  int co = (idx / T) % cout;
  int b  = idx / (T * cout);
  float acc = bias[co];
  for (int c = 0; c < cin; ++c) {
    const float* xr = X + ((size_t)b * cinTotal + c) * T;
    const float* wr = W + ((size_t)co * cin + c) * 3;
#pragma unroll
    for (int kk = 0; kk < 3; ++kk) {
      int tt = t + kk - 1;
      if (tt >= 0 && tt < T) acc += xr[tt] * wr[kk];
    }
  }
  Y[idx] = fmaxf(acc, 0.0f);
}

// ---------- head: w3 1x1 conv, sigmoid(0.01*z) -> xc, xb_start, xb_end ----------
__global__ void k_head3(const float* __restrict__ H, const float* __restrict__ W3,
                        const float* __restrict__ b3, float* __restrict__ out) {
  int idx = blockIdx.x * 256 + threadIdx.x;          // 4*128
  if (idx >= 512) return;
  int t = idx & 127, b = idx >> 7;
  float z0 = b3[0], z1 = b3[1], z2 = b3[2];
  for (int c = 0; c < 128; ++c) {
    float h = H[((size_t)b * 128 + c) * 128 + t];
    z0 += h * W3[c]; z1 += h * W3[128 + c]; z2 += h * W3[256 + c];
  }
  out[idx]        = sigmoidf(0.01f * z2);   // xc
  out[512 + idx]  = sigmoidf(0.01f * z0);   // xb_start
  out[1024 + idx] = sigmoidf(0.01f * z1);   // xb_end
}

// ---------- G[(o*4+b)][t*32+s] = sum_c h[b,c,t]*w3d[o,c,s]  (bf16 out) ----------
__global__ void k_build_G(const float* __restrict__ H, const float* __restrict__ W3d,
                          __bf16* __restrict__ G) {
  int idx = blockIdx.x * 256 + threadIdx.x;          // 2048*4096 exactly
  int s = idx & 31;
  int t = (idx >> 5) & 127;
  int b = (idx >> 12) & 3;
  int o = idx >> 14;
  const float* hp = H   + ((size_t)b * 128) * 128 + t;   // stride 128 over c
  const float* wp = W3d + ((size_t)o * 128) * 32  + s;   // stride 32  over c
  float acc = 0.0f;
  for (int c = 0; c < 128; ++c) acc += hp[c * 128] * wp[c * 32];
  G[idx] = f32_to_bf16(acc);
}

// ---------- bf16 WMMA GEMM: C = relu(A@B + bias), bf16 out ----------
// A: MxK row-major bf16 (TDM-staged); B: KxN row-major bf16; C: MxN row-major bf16
// Requires: M % 128 == 0 (or M == BM), N % 128 == 0, K % 32 == 0.
// bias index for row m: bias[(m / bias_div) % bias_mod]
#define BM 128
#define BN 128
#define BK 32
__global__ __launch_bounds__(256) void k_gemm_bf16(
    const __bf16* __restrict__ A, const __bf16* __restrict__ Bm,
    __bf16* __restrict__ C, const float* __restrict__ bias,
    int bias_div, int bias_mod, int M, int N, int K)
{
  // padded rows: 40 bf16 = 80B (16B-aligned, 20-bank stride kills ds conflicts)
  __shared__ __attribute__((aligned(16))) __bf16 As[BM][BK + 8];
  __shared__ __attribute__((aligned(16))) __bf16 Bs[BN][BK + 8];  // transposed: [n][k]

  const int tid  = threadIdx.x;
  const int lane = tid & 31;
  const int wave = tid >> 5;
  const int wm   = (wave >> 1) * 32;     // 4 waves along M (32 each)
  const int wn   = (wave & 1) * 64;      // 2 waves along N (64 each)
  const int half = lane >> 4;
  const int l16  = lane & 15;
  const int m0   = blockIdx.y * BM;
  const int n0   = blockIdx.x * BN;

  v8f acc[2][4] = {};

  // --- TDM descriptor, loop-invariant part (group1; groups 2/3 zero for 2D) ---
  // data_size=2B(code 1), pad_enable, pad_interval=16 dwords(code 3), pad_amount=4 dwords(code 3)
  const unsigned lds_a = (unsigned)(unsigned long long)(uintptr_t)&As[0][0];
  i32x8 g1;
  g1[0] = (int)((1u << 16) | (1u << 20) | (3u << 22) | (3u << 25));
  g1[1] = (int)(((unsigned)K & 0xFFFFu) << 16);                                   // tensor_dim0 lo
  g1[2] = (int)((((unsigned)K >> 16) & 0xFFFFu) | (((unsigned)M & 0xFFFFu) << 16)); // dim0 hi | dim1 lo
  g1[3] = (int)((((unsigned)M >> 16) & 0xFFFFu) | ((unsigned)BK << 16));          // dim1 hi | tile_dim0=32
  g1[4] = (int)(BM & 0xFFFF);                                                     // tile_dim1=128, tile_dim2=0
  g1[5] = (int)K;                                                                 // tensor_dim0_stride lo
  g1[6] = 0;
  g1[7] = 0;
  const i32x4 gz4 = {0, 0, 0, 0};
  const i32x8 gz8 = {0, 0, 0, 0, 0, 0, 0, 0};

  for (int k0 = 0; k0 < K; k0 += BK) {
    __syncthreads();  // previous compute done before LDS overwrite
    if (tid < 32) {
      // wave 0: TDM-load the A tile (128x32, padded to 40-elem rows) into LDS
      unsigned long long ga = (unsigned long long)(uintptr_t)A +
                              ((unsigned long long)m0 * (unsigned)K + (unsigned)k0) * 2ull;
      u32x4 g0;
      g0[0] = 1u;                                                  // count=1, user descriptor
      g0[1] = lds_a;                                               // LDS byte address
      g0[2] = (unsigned)ga;                                        // global addr lo
      g0[3] = (unsigned)((ga >> 32) & 0x01FFFFFFu) | (2u << 30);   // addr hi | type=2
      __builtin_amdgcn_tensor_load_to_lds(g0, g1, gz4, gz4, gz8, 0);
    }
    // all waves: stage B tile (BK x BN) — 16B coalesced global loads, transpose into Bs[n][k]
    for (int c = tid; c < (BK * BN) / 8; c += 256) {
      int k = c >> 4;             // 16 chunks of 8 elements along N per k-row
      int n = (c & 15) * 8;
      const v8bf v = *(const v8bf*)&Bm[(size_t)(k0 + k) * N + n0 + n];
#pragma unroll
      for (int e = 0; e < 8; ++e) Bs[n + e][k] = v[e];
    }
    if (k0 + BK < K)  // hint next B tile toward L2 (global_prefetch_b8)
      __builtin_prefetch(&Bm[(size_t)(k0 + BK) * N + n0 + (tid & 127)], 0, 1);
    if (tid < 32) __builtin_amdgcn_s_wait_tensorcnt(0);
    __syncthreads();  // As (TDM) + Bs (stores) visible to all waves

    // A fragment: lane (half,l16) -> row m, k = e%8 + 8*half + 16*(e/8)
    v16bf aF[2], bF[4];
#pragma unroll
    for (int i = 0; i < 2; ++i) {
      const v8bf lo = *(const v8bf*)&As[wm + i * 16 + l16][8 * half];
      const v8bf hi = *(const v8bf*)&As[wm + i * 16 + l16][16 + 8 * half];
#pragma unroll
      for (int e = 0; e < 8; ++e) { aF[i][e] = lo[e]; aF[i][e + 8] = hi[e]; }
    }
    // B fragment: lane -> col n = l16, k = e%8 + 8*half + 16*(e/8)
#pragma unroll
    for (int j = 0; j < 4; ++j) {
      const v8bf lo = *(const v8bf*)&Bs[wn + j * 16 + l16][8 * half];
      const v8bf hi = *(const v8bf*)&Bs[wn + j * 16 + l16][16 + 8 * half];
#pragma unroll
      for (int e = 0; e < 8; ++e) { bF[j][e] = lo[e]; bF[j][e + 8] = hi[e]; }
    }
#pragma unroll
    for (int i = 0; i < 2; ++i)
#pragma unroll
      for (int j = 0; j < 4; ++j)
        acc[i][j] = __builtin_amdgcn_wmma_f32_16x16x32_bf16(
            false, aF[i], false, bF[j], (short)0, acc[i][j], false, false);
  }

  // epilogue: D layout — lane l, vgpr v: n = l%16, m = v + 8*(l/16)
#pragma unroll
  for (int i = 0; i < 2; ++i) {
#pragma unroll
    for (int j = 0; j < 4; ++j) {
      int gn = n0 + wn + j * 16 + l16;
#pragma unroll
      for (int v = 0; v < 8; ++v) {
        int gm = m0 + wm + i * 16 + v + 8 * half;
        if (gm < M && gn < N) {
          float val = acc[i][j][v] + bias[(gm / bias_div) % bias_mod];
          C[(size_t)gm * N + gn] = f32_to_bf16(fmaxf(val, 0.0f));
        }
      }
    }
  }
}

// ---------- direct 3x3 conv (pad=1) + ReLU, layout (C,B,H,W), f32 out ----------
__device__ __forceinline__ float ldf(const float* p)  { return *p; }
__device__ __forceinline__ float ldf(const __bf16* p) { return bf16_to_f32(*p); }

template <typename Tin>
__global__ void k_conv3x3_relu(const Tin* __restrict__ X, const float* __restrict__ W,
                               const float* __restrict__ bias, float* __restrict__ Y, int Cin) {
  int idx = blockIdx.x * 256 + threadIdx.x;          // exactly Cout*4*128*128
  int j  = idx & 127;
  int i  = (idx >> 7) & 127;
  int b  = (idx >> 14) & 3;
  int co = idx >> 16;
  float acc = bias[co];
  for (int c = 0; c < Cin; ++c) {
    const Tin*   xb = X + ((size_t)c * 4 + b) * 16384;
    const float* wr = W + ((size_t)co * Cin + c) * 9;
#pragma unroll
    for (int di = 0; di < 3; ++di) {
      int ii = i + di - 1;
      if (ii < 0 || ii > 127) continue;
#pragma unroll
      for (int dj = 0; dj < 3; ++dj) {
        int jj = j + dj - 1;
        if (jj < 0 || jj > 127) continue;
        acc += ldf(&xb[ii * 128 + jj]) * wr[di * 3 + dj];
      }
    }
  }
  Y[((size_t)co * 4 + b) * 16384 + i * 128 + j] = fmaxf(acc, 0.0f);
}

// ---------- head: wd 1x1 conv + sigmoid -> iou, prop_start, prop_end ----------
__global__ void k_head4(const float* __restrict__ P, const float* __restrict__ Wd,
                        const float* __restrict__ bd, float* __restrict__ out) {
  int idx = blockIdx.x * 256 + threadIdx.x;          // 4*16384
  int ij = idx & 16383, b = idx >> 14;
  float z[4];
#pragma unroll
  for (int oc = 0; oc < 4; ++oc) z[oc] = bd[oc];
  for (int c = 0; c < 64; ++c) {
    float x = P[((size_t)c * 4 + b) * 16384 + ij];
#pragma unroll
    for (int oc = 0; oc < 4; ++oc) z[oc] += x * Wd[oc * 64 + c];
  }
#pragma unroll
  for (int oc = 0; oc < 4; ++oc) z[oc] = sigmoidf(z[oc]);
  out[1536 + ((size_t)b * 2 + 0) * 16384 + ij]        = z[2];  // iou ch0
  out[1536 + ((size_t)b * 2 + 1) * 16384 + ij]        = z[3];  // iou ch1
  out[1536 + 131072 + (size_t)b * 16384 + ij]         = z[0];  // prop_start
  out[1536 + 131072 + 65536 + (size_t)b * 16384 + ij] = z[1];  // prop_end
}

extern "C" void kernel_launch(void* const* d_in, const int* in_sizes, int n_in,
                              void* d_out, int out_size, void* d_ws, size_t ws_size,
                              hipStream_t stream) {
  (void)in_sizes; (void)n_in; (void)out_size; (void)ws_size;
  const float* x    = (const float*)d_in[0];
  const float* w1   = (const float*)d_in[1];
  const float* b1   = (const float*)d_in[2];
  const float* w2   = (const float*)d_in[3];
  const float* b2   = (const float*)d_in[4];
  const float* w3   = (const float*)d_in[5];
  const float* b3   = (const float*)d_in[6];
  const float* w3d  = (const float*)d_in[7];
  const float* b3d  = (const float*)d_in[8];
  const float* wa   = (const float*)d_in[9];
  const float* ba   = (const float*)d_in[10];
  const float* wb   = (const float*)d_in[11];
  const float* bb   = (const float*)d_in[12];
  const float* wc   = (const float*)d_in[13];
  const float* bc   = (const float*)d_in[14];
  const float* wd   = (const float*)d_in[15];
  const float* bd   = (const float*)d_in[16];
  const float* mask = (const float*)d_in[17];
  float* out = (float*)d_out;

  char* wsp = (char*)d_ws;
  auto alloc = [&](size_t bytes) -> char* {
    char* p = wsp; wsp += (bytes + 255) & ~(size_t)255; return p;
  };
  __bf16* mask_bf = (__bf16*)alloc(67108864ull * 2);   // 4096 x 16384
  float*  h1      = (float*) alloc(262144ull * 4);     // (4,512,128)
  float*  h       = (float*) alloc(65536ull * 4);      // (4,128,128)
  __bf16* G       = (__bf16*)alloc(8388608ull * 2);    // 2048 x 4096
  __bf16* cm      = (__bf16*)alloc(33554432ull * 2);   // 512 x 65536, (o,b,i,j)
  __bf16* wa_bf   = (__bf16*)alloc(65536ull * 2);      // 128 x 512
  __bf16* p1      = (__bf16*)alloc(8388608ull * 2);    // 128 x 65536, (c,b,i,j)
  float*  p2      = (float*) alloc(4194304ull * 4);    // 64  x 65536
  float*  p3      = (float*) alloc(4194304ull * 4);    // 64  x 65536

  // 1) convert the 268MB mask to bf16 once
  k_cast_bf16<<<262144, 256, 0, stream>>>(mask, mask_bf, 67108864);
  // 2) prologue convs + boundary heads
  k_conv1d_relu<<<1024, 256, 0, stream>>>(x, 400, 200, w1, b1, h1, 512);
  k_conv1d_relu<<<256,  256, 0, stream>>>(h1, 512, 512, w2, b2, h, 128);
  k_head3<<<2, 256, 0, stream>>>(h, w3, b3, out);
  // 3) fold w3d into the T-contraction: G = h x w3d  (2.1 GFLOP)
  k_build_G<<<32768, 256, 0, stream>>>(h, w3d, G);
  // 4) the big WMMA GEMM: cm = relu(G @ mask + b3d)  (2048x16384x4096, 274 GFLOP)
  dim3 gBig(16384 / BN, 2048 / BM);
  k_gemm_bf16<<<gBig, 256, 0, stream>>>(G, mask_bf, cm, b3d, 4, 512, 2048, 16384, 4096);
  // 5) 1x1 conv wa as WMMA GEMM: p1 = relu(wa @ cm + ba)  (128x65536x512)
  k_cast_bf16<<<256, 256, 0, stream>>>(wa, wa_bf, 65536);
  dim3 gWa(65536 / BN, 1);
  k_gemm_bf16<<<gWa, 256, 0, stream>>>(wa_bf, cm, p1, ba, 1, 128, 128, 65536, 512);
  // 6) 3x3 convs (direct f32, inputs L2-resident at 16MB)
  k_conv3x3_relu<__bf16><<<16384, 256, 0, stream>>>(p1, wb, bb, p2, 128);
  k_conv3x3_relu<float> <<<16384, 256, 0, stream>>>(p2, wc, bc, p3, 64);
  // 7) final 1x1 head + sigmoid
  k_head4<<<256, 256, 0, stream>>>(p3, wd, bd, out);
}
